// GraphAttentionAutoEncoder_50371376447824
// MI455X (gfx1250) — matrix operand order repaired and checked by reading, not compiled, not run on hardware
//
#include <hip/hip_runtime.h>
#include <hip/hip_bf16.h>

// ---------------------------------------------------------------------------
// Graph U-Net (GAT + TopKPool) for MI455X / gfx1250, wave32, WMMA f16->f32.
// ---------------------------------------------------------------------------

typedef __attribute__((ext_vector_type(16))) _Float16 v16h;
typedef __attribute__((ext_vector_type(8)))  float    v8f;

#define E0 800000
#define NEG_SLOPE 0.2f

#define GS_LOOP(i, total) \
  for (long i = blockIdx.x * (long)blockDim.x + threadIdx.x; i < (total); \
       i += (long)gridDim.x * blockDim.x)

static __device__ __forceinline__ void atomic_add_f32(float* p, float v) {
  (void)__hip_atomic_fetch_add(p, v, __ATOMIC_RELAXED, __HIP_MEMORY_SCOPE_AGENT);
}

static __device__ __forceinline__ void atomic_max_f32(float* p, float v) {
  int* ai = (int*)p;
  int old = *ai;
  while (__int_as_float(old) < v) {
    int assumed = old;
    old = atomicCAS(ai, assumed, __float_as_int(v));
    if (old == assumed) break;
  }
}

// ---------------------------------------------------------------------------
// WMMA GEMM: XL[N, Cout] = X[N, Cin] @ W[Cout, Cin]^T   (f16 in, f32 acc)
// One block = 4 waves; block owns 16 rows; wave w owns columns [16w,16w+16).
// A and B are staged in LDS pre-swizzled into per-lane WMMA fragment order so
// each lane's 16-half fragment is one aligned 32B chunk (2x ds_load_b128).
// ---------------------------------------------------------------------------
__global__ __launch_bounds__(128)
void k_linear_wmma(const float* __restrict__ X, const float* __restrict__ W,
                   float* __restrict__ XL, int N, int Cin, int Cout, int Kpad) {
  // Ash: [Kpad/32][16 rows][32 halves]   (fragment-ordered)
  // Bsh: [Kpad/32][64 cols][32 halves]   (fragment-ordered)
  __shared__ __align__(32) _Float16 Ash[16 * 128];
  __shared__ __align__(32) _Float16 Bsh[64 * 128];
  const int tid   = threadIdx.x;
  const int mbase = blockIdx.x * 16;

  // Stage A swizzled: per k-block, per row: order K = {0..7, 16..23, 8..15, 24..31}
  for (int idx = tid; idx < 16 * Kpad; idx += 128) {
    int r = idx / Kpad, k = idx - r * Kpad;
    int row = mbase + r;
    float v = (row < N && k < Cin) ? X[(long)row * Cin + k] : 0.f;
    int kb32 = k >> 5, kr = k & 31;
    int pos = (kr & 7) + ((kr & 16) ? 8 : 0) + ((kr & 8) ? 16 : 0);
    Ash[((kb32 * 16) + r) * 32 + pos] = (_Float16)v;
  }
  // Stage B swizzled: per k-block, per col: order K = {0..7, 16..23, 8..15, 24..31}
  for (int idx = tid; idx < Kpad * 64; idx += 128) {
    int k = idx >> 6, o = idx & 63;
    float v = (k < Cin && o < Cout) ? W[o * Cin + k] : 0.f;
    int kb32 = k >> 5, kr = k & 31;
    Bsh[((kb32 << 6) + o) * 32 + (kr & 7) + ((kr & 16) ? 8 : 0) + ((kr & 8) ? 16 : 0)] = (_Float16)v;
  }
  __syncthreads();

  const int wave  = tid >> 5;
  const int lane  = tid & 31;
  const int nbase = wave * 16;
  if (nbase < Cout) {                       // wave-uniform branch (EXEC all-1)
    v8f acc = {0.f, 0.f, 0.f, 0.f, 0.f, 0.f, 0.f, 0.f};
    const int m     = lane & 15;
    const int half16 = (lane < 16) ? 0 : 16; // fragment half within 32-half chunk
    const int ncol  = nbase + (lane & 15);
    const int nkb   = Kpad >> 5;
    for (int kb32 = 0; kb32 < nkb; ++kb32) {
      v16h a = *(const v16h*)(Ash + ((kb32 * 16) + m) * 32 + half16);
      v16h b = *(const v16h*)(Bsh + ((kb32 * 64) + ncol) * 32 + half16);
      acc = __builtin_amdgcn_wmma_f32_16x16x32_f16(
          false, a, false, b, (short)0, acc, false, false);
    }
#pragma unroll
    for (int r = 0; r < 8; ++r) {
      int row = mbase + ((lane < 16) ? r : (r + 8));
      if (row < N && ncol < Cout) XL[(long)row * Cout + ncol] = acc[r];
    }
  }
}

// ---------------------------------------------------------------------------
// Attention logits: aS[n,h] = sum_c XL[n,h,c]*attS[h,c]  (same for aD)
// ---------------------------------------------------------------------------
__global__ void k_logits(const float* __restrict__ XL, const float* __restrict__ attS,
                         const float* __restrict__ attD, float* aS, float* aD,
                         int N, int H, int C) {
  GS_LOOP(idx, (long)N * H) {
    long n = idx / H; int h = (int)(idx - n * H);
    const float* x = XL + n * (long)(H * C) + h * C;
    float s = 0.f, d = 0.f;
    for (int c = 0; c < C; ++c) { float v = x[c]; s += v * attS[h * C + c]; d += v * attD[h * C + c]; }
    aS[idx] = s; aD[idx] = d;
  }
}

__global__ void k_fill_f32(float* p, float v, long n) { GS_LOOP(i, n) p[i] = v; }
__global__ void k_fill_i32(int* p, int v, long n)     { GS_LOOP(i, n) p[i] = v; }

// ---------------------------------------------------------------------------
// Edge pass 1: segment max of leaky_relu(aS[s]+aD[d]) over dst (incl. self loops)
// ---------------------------------------------------------------------------
__global__ void k_edge_max(const int* __restrict__ src, const int* __restrict__ dst,
                           const int* __restrict__ mask, const float* __restrict__ aS,
                           const float* __restrict__ aD, float* amax, int E, int N, int H) {
  GS_LOOP(idx, (long)(E + N) * H) {
    long e = idx / H; int h = (int)(idx - e * H);
    int s, d;
    if (e < E) { if (mask && !mask[e]) continue; s = src[e]; d = dst[e]; }
    else       { s = d = (int)(e - E); }
    float al = aS[(long)s * H + h] + aD[(long)d * H + h];
    al = al > 0.f ? al : NEG_SLOPE * al;
    atomic_max_f32(&amax[(long)d * H + h], al);
  }
}

// Edge pass 2: expv = exp(alpha - amax[d]); denom[d] += expv
__global__ void k_edge_exps(const int* __restrict__ src, const int* __restrict__ dst,
                            const int* __restrict__ mask, const float* __restrict__ aS,
                            const float* __restrict__ aD, const float* __restrict__ amax,
                            float* __restrict__ expv, float* denom, int E, int N, int H) {
  GS_LOOP(idx, (long)(E + N) * H) {
    long e = idx / H; int h = (int)(idx - e * H);
    int s, d;
    if (e < E) {
      if (mask && !mask[e]) { expv[idx] = 0.f; continue; }
      s = src[e]; d = dst[e];
    } else { s = d = (int)(e - E); }
    float al = aS[(long)s * H + h] + aD[(long)d * H + h];
    al = al > 0.f ? al : NEG_SLOPE * al;
    float v = __expf(al - amax[(long)d * H + h]);
    expv[idx] = v;
    atomic_add_f32(&denom[(long)d * H + h], v);
  }
}

// Edge pass 3: OUT[d,h,c] += (expv/denom[d]) * XL[s,h,c]
__global__ void k_edge_scatter(const int* __restrict__ src, const int* __restrict__ dst,
                               const int* __restrict__ mask, const float* __restrict__ expv,
                               const float* __restrict__ denom, const float* __restrict__ XL,
                               float* OUT, int E, int N, int H, int C) {
  const int HC = H * C;
  GS_LOOP(idx, (long)(E + N) * HC) {
    long e = idx / HC; int j = (int)(idx - e * HC); int h = j / C;
    int s, d;
    if (e < E) { if (mask && !mask[e]) continue; s = src[e]; d = dst[e]; }
    else       { s = d = (int)(e - E); }
    const float* xs = XL + (long)s * HC + j;
    __builtin_prefetch((const void*)xs, 0, 1);  // global_prefetch_b8 (gather side)
    float w = expv[e * H + h];
    if (w != 0.f) {
      w /= denom[(long)d * H + h];
      atomic_add_f32(&OUT[(long)d * HC + j], w * (*xs));
    }
  }
}

__global__ void k_bias_act(float* OUT, const float* __restrict__ b, long total, int HC, int do_elu) {
  GS_LOOP(idx, total) {
    int j = (int)(idx % HC);
    float v = OUT[idx] + b[j];
    if (do_elu) v = v > 0.f ? v : expm1f(v);
    OUT[idx] = v;
  }
}

// ---------------------------------------------------------------------------
// TopK pooling
// ---------------------------------------------------------------------------
__global__ void k_score(const float* __restrict__ X, const float* __restrict__ w,
                        float* score, unsigned* keys, int N, int C) {
  GS_LOOP(n, (long)N) {
    float nrm = 0.f, dot = 0.f;
    for (int c = 0; c < C; ++c) { float wc = w[c]; nrm += wc * wc; dot += X[n * C + c] * wc; }
    float sc = tanhf(dot * rsqrtf(nrm));
    score[n] = sc;
    unsigned u = __float_as_uint(sc);
    keys[n] = (u & 0x80000000u) ? ~u : (u | 0x80000000u);
  }
}

// Radix bisection: sel[0]=threshold key (k-th largest), sel[1]=count(keys>T)
__global__ __launch_bounds__(1024)
void k_select(const unsigned* __restrict__ keys, int N, int k, unsigned* sel) {
  __shared__ int red[1024];
  __shared__ unsigned spfx;
  const int tid = threadIdx.x;
  if (tid == 0) spfx = 0u;
  __syncthreads();
  for (int bit = 31; bit >= 0; --bit) {
    unsigned cand = spfx | (1u << bit);
    int cnt = 0;
    for (int i = tid; i < N; i += 1024) cnt += (keys[i] >= cand);
    red[tid] = cnt; __syncthreads();
    for (int s = 512; s > 0; s >>= 1) { if (tid < s) red[tid] += red[tid + s]; __syncthreads(); }
    if (tid == 0 && red[0] >= k) spfx = cand;
    __syncthreads();
  }
  unsigned T = spfx;
  int cnt = 0;
  for (int i = tid; i < N; i += 1024) cnt += (keys[i] > T);
  red[tid] = cnt; __syncthreads();
  for (int s = 512; s > 0; s >>= 1) { if (tid < s) red[tid] += red[tid + s]; __syncthreads(); }
  if (tid == 0) { sel[0] = T; sel[1] = (unsigned)red[0]; }
}

__global__ void k_part_count(const unsigned* __restrict__ keys, int N,
                             const unsigned* __restrict__ sel, int* bG, int* bE) {
  __shared__ int sg[256], se[256];
  const unsigned T = sel[0];
  int n = blockIdx.x * 256 + threadIdx.x;
  int g = 0, e = 0;
  if (n < N) { unsigned ky = keys[n]; g = ky > T; e = ky == T; }
  sg[threadIdx.x] = g; se[threadIdx.x] = e; __syncthreads();
  for (int s = 128; s > 0; s >>= 1) {
    if (threadIdx.x < s) { sg[threadIdx.x] += sg[threadIdx.x + s]; se[threadIdx.x] += se[threadIdx.x + s]; }
    __syncthreads();
  }
  if (threadIdx.x == 0) { bG[blockIdx.x] = sg[0]; bE[blockIdx.x] = se[0]; }
}

__global__ void k_scan_blocks(int* bG, int* bE, int nb) {
  if (blockIdx.x == 0 && threadIdx.x == 0) {
    int aG = 0, aE = 0;
    for (int i = 0; i < nb; ++i) { int g = bG[i], e = bE[i]; bG[i] = aG; bE[i] = aE; aG += g; aE += e; }
  }
}

__global__ void k_scatter_perm(const unsigned* __restrict__ keys, const unsigned* __restrict__ sel,
                               const int* __restrict__ bG, const int* __restrict__ bE,
                               int* perm, int N, int k) {
  __shared__ int sg[256], se[256];
  const unsigned T = sel[0];
  const int gtot = (int)sel[1];
  const int tid = threadIdx.x;
  int n = blockIdx.x * 256 + tid;
  int g = 0, e = 0;
  if (n < N) { unsigned ky = keys[n]; g = ky > T; e = (ky == T); }
  sg[tid] = g; se[tid] = e; __syncthreads();
  for (int off = 1; off < 256; off <<= 1) {           // inclusive scan
    int vg = (tid >= off) ? sg[tid - off] : 0;
    int ve = (tid >= off) ? se[tid - off] : 0;
    __syncthreads();
    sg[tid] += vg; se[tid] += ve;
    __syncthreads();
  }
  if (n < N) {
    if (g)      { int p = bG[blockIdx.x] + sg[tid] - 1;        if (p < k) perm[p] = n; }
    else if (e) { int p = gtot + bE[blockIdx.x] + se[tid] - 1; if (p < k) perm[p] = n; }
  }
}

__global__ void k_gather_gate(const float* __restrict__ X, const int* __restrict__ perm,
                              const float* __restrict__ score, float* Xn, int k, int C) {
  GS_LOOP(idx, (long)k * C) {
    long i = idx / C; int c = (int)(idx - i * C);
    int n = perm[i];
    Xn[idx] = X[(long)n * C + c] * score[n];
  }
}

__global__ void k_set_cluster(const int* __restrict__ perm, int* cluster, int k) {
  GS_LOOP(i, (long)k) cluster[perm[i]] = (int)i;
}

__global__ void k_relabel(const int* __restrict__ src, const int* __restrict__ dst,
                          const int* __restrict__ mask, const int* __restrict__ cluster,
                          int* ns, int* nd, int* nm, int E) {
  GS_LOOP(e, (long)E) {
    int a = cluster[src[e]], b = cluster[dst[e]];
    int m = (mask ? mask[e] : 1) && a >= 0 && b >= 0;
    nm[e] = m; ns[e] = m ? a : 0; nd[e] = m ? b : 0;
  }
}

// ---------------------------------------------------------------------------
// Decoder helpers: concat / unpool-scatter
// ---------------------------------------------------------------------------
__global__ void k_concat2(const float* __restrict__ A, const float* __restrict__ B,
                          float* Xcat, int N) {
  GS_LOOP(idx, (long)N * 128) {
    long n = idx >> 7; int c = (int)(idx & 127);
    Xcat[idx] = (c < 64) ? A[n * 64 + c] : B[n * 64 + (c - 64)];
  }
}
__global__ void k_initcat(float* Xcat, const float* __restrict__ Xenc, int N) {
  GS_LOOP(idx, (long)N * 128) {
    long n = idx >> 7; int c = (int)(idx & 127);
    Xcat[idx] = (c < 64) ? 0.f : Xenc[n * 64 + (c - 64)];
  }
}
__global__ void k_scatter_unpool(const float* __restrict__ Xs, const int* __restrict__ perm,
                                 float* Xcat, int k) {
  GS_LOOP(idx, (long)k * 64) {
    long i = idx >> 6; int c = (int)(idx & 63);
    Xcat[(long)perm[i] * 128 + c] = Xs[idx];
  }
}

// ---------------------------------------------------------------------------
// Host orchestration
// ---------------------------------------------------------------------------
static inline unsigned grid_for(long total) {
  long g = (total + 255) / 256;
  if (g < 1) g = 1;
  if (g > 262144) g = 262144;
  return (unsigned)g;
}

struct GatScratch { float *XL, *aS, *aD, *amax, *denom, *expv; };

static void launch_gat(hipStream_t st, const float* X, int N, int Cin,
                       const int* src, const int* dst, const int* mask,
                       const float* W, const float* attS, const float* attD,
                       const float* bias, int H, int C, int do_elu,
                       float* OUT, const GatScratch& s) {
  const int Kpad = ((Cin + 31) / 32) * 32;
  const int HC = H * C;
  k_linear_wmma<<<(N + 15) / 16, 128, 0, st>>>(X, W, s.XL, N, Cin, HC, Kpad);
  const long tNH = (long)N * H;
  k_logits<<<grid_for(tNH), 256, 0, st>>>(s.XL, attS, attD, s.aS, s.aD, N, H, C);
  k_fill_f32<<<grid_for(tNH), 256, 0, st>>>(s.amax, -__builtin_inff(), tNH);
  k_fill_f32<<<grid_for(tNH), 256, 0, st>>>(s.denom, 0.f, tNH);
  k_fill_f32<<<grid_for((long)N * HC), 256, 0, st>>>(OUT, 0.f, (long)N * HC);
  const long tEH = (long)(E0 + N) * H;
  k_edge_max<<<grid_for(tEH), 256, 0, st>>>(src, dst, mask, s.aS, s.aD, s.amax, E0, N, H);
  k_edge_exps<<<grid_for(tEH), 256, 0, st>>>(src, dst, mask, s.aS, s.aD, s.amax, s.expv, s.denom, E0, N, H);
  k_edge_scatter<<<grid_for((long)(E0 + N) * HC), 256, 0, st>>>(src, dst, mask, s.expv, s.denom, s.XL, OUT, E0, N, H, C);
  k_bias_act<<<grid_for((long)N * HC), 256, 0, st>>>(OUT, bias, (long)N * HC, HC, do_elu);
}

extern "C" void kernel_launch(void* const* d_in, const int* in_sizes, int n_in,
                              void* d_out, int out_size, void* d_ws, size_t ws_size,
                              hipStream_t stream) {
  (void)in_sizes; (void)n_in; (void)out_size; (void)ws_size;
  static const int NS[4] = {100000, 50000, 25000, 12500};

  const float* x_in = (const float*)d_in[0];
  const int*   eidx = (const int*)d_in[1];
  // d_in[2] = batch (all zeros, unused)
  const float *encW[4], *encAs[4], *encAd[4], *encB[4];
  const float *poolw[3];
  const float *decW[4], *decAs[4], *decAd[4], *decB[4];
  int p = 3;
  for (int i = 0; i < 4; ++i) { encW[i] = (const float*)d_in[p++]; encAs[i] = (const float*)d_in[p++];
                                encAd[i] = (const float*)d_in[p++]; encB[i] = (const float*)d_in[p++]; }
  for (int i = 0; i < 3; ++i) poolw[i] = (const float*)d_in[p++];
  for (int i = 0; i < 4; ++i) { decW[i] = (const float*)d_in[p++]; decAs[i] = (const float*)d_in[p++];
                                decAd[i] = (const float*)d_in[p++]; decB[i] = (const float*)d_in[p++]; }

  // bump allocator over workspace
  char* ws = (char*)d_ws; size_t off = 0;
  auto alloc = [&](size_t bytes) -> void* {
    size_t a = (off + 255) & ~(size_t)255; void* r = ws + a; off = a + bytes; return r;
  };
  const int N0 = NS[0];
  float* xenc[4]; for (int i = 0; i < 4; ++i) xenc[i] = (float*)alloc((size_t)NS[i] * 64 * 4);
  GatScratch scr;
  scr.XL    = (float*)alloc((size_t)N0 * 64 * 4);
  scr.aS    = (float*)alloc((size_t)N0 * 4 * 4);
  scr.aD    = (float*)alloc((size_t)N0 * 4 * 4);
  scr.amax  = (float*)alloc((size_t)N0 * 4 * 4);
  scr.denom = (float*)alloc((size_t)N0 * 4 * 4);
  scr.expv  = (float*)alloc((size_t)(E0 + N0) * 4 * 4);
  float*    xpool  = (float*)alloc((size_t)NS[1] * 64 * 4);
  float*    xcat   = (float*)alloc((size_t)N0 * 128 * 4);
  float*    xdec   = (float*)alloc((size_t)N0 * 64 * 4);
  float*    score  = (float*)alloc((size_t)N0 * 4);
  unsigned* keys   = (unsigned*)alloc((size_t)N0 * 4);
  unsigned* sel    = (unsigned*)alloc(64);
  int*      bG     = (int*)alloc(4096);
  int*      bE     = (int*)alloc(4096);
  int*      cluster= (int*)alloc((size_t)N0 * 4);
  int* permA[3];  for (int i = 0; i < 3; ++i) permA[i] = (int*)alloc((size_t)NS[i + 1] * 4);
  const int *srcL[4], *dstL[4], *maskL[4];
  srcL[0] = eidx; dstL[0] = eidx + E0; maskL[0] = nullptr;
  for (int l = 1; l < 4; ++l) {
    srcL[l]  = (int*)alloc((size_t)E0 * 4);
    dstL[l]  = (int*)alloc((size_t)E0 * 4);
    maskL[l] = (int*)alloc((size_t)E0 * 4);
  }

  // ---------------- Encoder ----------------
  for (int i = 0; i < 4; ++i) {
    const int N = NS[i];
    const int Cin = (i == 0) ? 6 : 64;
    const float* X = (i == 0) ? x_in : xpool;
    launch_gat(stream, X, N, Cin, srcL[i], dstL[i], maskL[i],
               encW[i], encAs[i], encAd[i], encB[i], 4, 16, 1, xenc[i], scr);
    if (i < 3) {
      const int k = NS[i + 1];
      const int nb = (N + 255) / 256;
      k_score<<<grid_for(N), 256, 0, stream>>>(xenc[i], poolw[i], score, keys, N, 64);
      k_select<<<1, 1024, 0, stream>>>(keys, N, k, sel);
      k_part_count<<<nb, 256, 0, stream>>>(keys, N, sel, bG, bE);
      k_scan_blocks<<<1, 1, 0, stream>>>(bG, bE, nb);
      k_scatter_perm<<<nb, 256, 0, stream>>>(keys, sel, bG, bE, permA[i], N, k);
      k_gather_gate<<<grid_for((long)k * 64), 256, 0, stream>>>(xenc[i], permA[i], score, xpool, k, 64);
      k_fill_i32<<<grid_for(N), 256, 0, stream>>>(cluster, -1, N);
      k_set_cluster<<<grid_for(k), 256, 0, stream>>>(permA[i], cluster, k);
      k_relabel<<<grid_for(E0), 256, 0, stream>>>(srcL[i], dstL[i], maskL[i], cluster,
                                                  (int*)srcL[i + 1], (int*)dstL[i + 1],
                                                  (int*)maskL[i + 1], E0);
    }
  }

  // ---------------- Decoder ----------------
  // step 0: idx=3, no unpool: x = xenc[3] concat xenc[3]
  k_concat2<<<grid_for((long)NS[3] * 128), 256, 0, stream>>>(xenc[3], xenc[3], xcat, NS[3]);
  launch_gat(stream, xcat, NS[3], 128, srcL[3], dstL[3], maskL[3],
             decW[0], decAs[0], decAd[0], decB[0], 4, 16, 1, xdec, scr);
  // steps 1..3: unpool via perm[idx], concat skip, GAT
  for (int i = 1; i < 4; ++i) {
    const int idx = 3 - i;
    const int Nprev = NS[idx];
    const int ksmall = NS[idx + 1];
    k_initcat<<<grid_for((long)Nprev * 128), 256, 0, stream>>>(xcat, xenc[idx], Nprev);
    k_scatter_unpool<<<grid_for((long)ksmall * 64), 256, 0, stream>>>(xdec, permA[idx], xcat, ksmall);
    if (i < 3) {
      launch_gat(stream, xcat, Nprev, 128, srcL[idx], dstL[idx], maskL[idx],
                 decW[i], decAs[i], decAd[i], decB[i], 4, 16, 1, xdec, scr);
    } else {
      // final layer: heads=1, cout=6, concat=False (mean of 1 head == itself), no ELU
      launch_gat(stream, xcat, Nprev, 128, srcL[idx], dstL[idx], maskL[idx],
                 decW[3], decAs[3], decAd[3], decB[3], 1, 6, 0, (float*)d_out, scr);
    }
  }
}
